// HConstructor_20710332302026
// MI455X (gfx1250) — matrix operand order repaired and checked by reading, not compiled, not run on hardware
//
#include <hip/hip_runtime.h>
#include <hip/hip_bf16.h>

// ---------------------------------------------------------------------------
// Problem constants (from the reference): N=65536, D=256, NS=1024, HID=256
// ---------------------------------------------------------------------------
#define PN   65536
#define PD   256
#define PNS  1024
#define PHID 256
#define PSCALE 0.0625f   // 256^-0.5
#define PEPS 1e-8f

typedef __attribute__((ext_vector_type(16))) _Float16 v16h;
typedef __attribute__((ext_vector_type(8)))  _Float16 v8h;
typedef __attribute__((ext_vector_type(8)))  float    v8f;

union HFrag { v16h v; v8h h[2]; };

#define LDS_LD 40   // halves per LDS row (80 B): bank-conflict stagger, keeps 16B align

#if __has_builtin(__builtin_amdgcn_s_wait_asynccnt)
#define WAIT_ASYNC(n) __builtin_amdgcn_s_wait_asynccnt(n)
#else
#define WAIT_ASYNC(n) asm volatile("s_wait_asynccnt %0" :: "i"(n) : "memory")
#endif

// gfx1250 async global->LDS copy (ASYNCcnt-tracked, bypasses VGPRs).
// %0 = LDS byte address (wave-relative; low 32 bits of generic LDS pointer),
// %1 = 64-bit global address.  16 bytes per lane.
__device__ __forceinline__ void async_copy_b128(const _Float16* g, _Float16* l) {
  unsigned lds = (unsigned)(unsigned long long)l;
  asm volatile("global_load_async_to_lds_b128 %0, %1, off"
               :: "v"(lds), "v"((unsigned long long)g) : "memory");
}

__device__ __forceinline__ unsigned fkey(float f) {
  unsigned u = __float_as_uint(f);
  return (u & 0x80000000u) ? ~u : (u | 0x80000000u);  // order-preserving key
}

// ---------------------------------------------------------------------------
// Generic NT GEMM: C[M,N] = act(scale * A[M,K] @ Bt[N,K]^T + bias)
// A, Bt are f16 row-major over K.  f32 accumulation via v_wmma_f32_16x16x32_f16.
// Block 256 threads = 8 waves; 64x128 block tile; each wave owns a 32x32
// sub-tile (4 WMMA per 32-K step).  Double-buffered LDS filled with
// global_load_async_to_lds_b128, drained with s_wait_asynccnt.
// ---------------------------------------------------------------------------
template<bool RELU, bool TRANS>
__global__ __launch_bounds__(256)
void gemm_nt_wmma(const _Float16* __restrict__ A, const _Float16* __restrict__ Bt,
                  const float* __restrict__ bias, float scale,
                  float* __restrict__ Cf, _Float16* __restrict__ Ch,
                  int M, int N, int K, int ldc)
{
  __shared__ __align__(16) _Float16 sA[2][64 * LDS_LD];
  __shared__ __align__(16) _Float16 sB[2][128 * LDS_LD];

  const int tid  = threadIdx.x;
  const int lane = tid & 31;
  const int wave = tid >> 5;
  const int wm   = wave >> 2;          // 0..1  (M sub-block of 32)
  const int wn   = wave & 3;           // 0..3  (N sub-block of 32)
  const int m0   = blockIdx.y * 64;
  const int n0   = blockIdx.x * 128;

  // staging assignment: 16B chunk per async op
  const int sRow = tid >> 2;           // 0..63
  const int sSeg = (tid & 3) * 8;      // 0,8,16,24 halves

  const _Float16* gA  = &A [(size_t)(m0 + sRow)      * K + sSeg];
  const _Float16* gB0 = &Bt[(size_t)(n0 + sRow)      * K + sSeg];
  const _Float16* gB1 = &Bt[(size_t)(n0 + 64 + sRow) * K + sSeg];
  _Float16* lA[2]  = { &sA[0][sRow * LDS_LD + sSeg],        &sA[1][sRow * LDS_LD + sSeg] };
  _Float16* lB0[2] = { &sB[0][sRow * LDS_LD + sSeg],        &sB[1][sRow * LDS_LD + sSeg] };
  _Float16* lB1[2] = { &sB[0][(64 + sRow) * LDS_LD + sSeg], &sB[1][(64 + sRow) * LDS_LD + sSeg] };

  const int r   = lane & 15;
  const int lhi = lane >> 4;           // 0/1
  const int hk  = lhi << 3;            // A-frag K base: 0 or 8
  const int kb  = lhi << 3;            // B-frag K base: 0 or 8

  v8f acc00 = {}, acc01 = {}, acc10 = {}, acc11 = {};

  const int nk = K >> 5;
  // prologue: stage 0
  async_copy_b128(gA,  lA[0]);
  async_copy_b128(gB0, lB0[0]);
  async_copy_b128(gB1, lB1[0]);

  for (int i = 0; i < nk; ++i) {
    const int cur = i & 1;
    if (i + 1 < nk) {
      const size_t koff = (size_t)(i + 1) << 5;
      async_copy_b128(gA  + koff, lA [cur ^ 1]);
      async_copy_b128(gB0 + koff, lB0[cur ^ 1]);
      async_copy_b128(gB1 + koff, lB1[cur ^ 1]);
      WAIT_ASYNC(3);                   // 3 newest (next stage) may remain in flight
    } else {
      WAIT_ASYNC(0);
    }
    __syncthreads();                   // all waves' writes to buffer `cur` visible

    // B fragments: columns wn*32 + {0,16} + r ; K runs {kb..kb+7, kb+16..kb+23}
    HFrag b0, b1;
    {
      const _Float16* p = &sB[cur][(wn * 32 + r) * LDS_LD + kb];
      b0.h[0] = *(const v8h*)p;
      b0.h[1] = *(const v8h*)(p + 16);
      p += 16 * LDS_LD;
      b1.h[0] = *(const v8h*)p;
      b1.h[1] = *(const v8h*)(p + 16);
    }
    // A fragments: rows wm*32 + {0,16} + r ; K runs {hk..hk+7, hk+16..hk+23}
    HFrag a0, a1;
    {
      const _Float16* p = &sA[cur][(wm * 32 + r) * LDS_LD + hk];
      a0.h[0] = *(const v8h*)p;
      a0.h[1] = *(const v8h*)(p + 16);
      p += 16 * LDS_LD;
      a1.h[0] = *(const v8h*)p;
      a1.h[1] = *(const v8h*)(p + 16);
    }
    acc00 = __builtin_amdgcn_wmma_f32_16x16x32_f16(false, a0.v, false, b0.v, (short)0, acc00, false, false);
    acc01 = __builtin_amdgcn_wmma_f32_16x16x32_f16(false, a0.v, false, b1.v, (short)0, acc01, false, false);
    acc10 = __builtin_amdgcn_wmma_f32_16x16x32_f16(false, a1.v, false, b0.v, (short)0, acc10, false, false);
    acc11 = __builtin_amdgcn_wmma_f32_16x16x32_f16(false, a1.v, false, b1.v, (short)0, acc11, false, false);
    __syncthreads();                   // done reading buffer `cur`
  }

  // epilogue: C 16x16 v8f layout -> element (m = i + 8*lhi, n = lane&15)
  const int cn0 = n0 + wn * 32 + r;
  const int cn1 = cn0 + 16;
  const float bv0 = bias ? bias[cn0] : 0.0f;
  const float bv1 = bias ? bias[cn1] : 0.0f;
  #pragma unroll
  for (int i = 0; i < 8; ++i) {
    const int gm0 = m0 + wm * 32 + lhi * 8 + i;
    const int gm1 = gm0 + 16;
    float v00 = acc00[i] * scale + bv0;
    float v01 = acc01[i] * scale + bv1;
    float v10 = acc10[i] * scale + bv0;
    float v11 = acc11[i] * scale + bv1;
    if (RELU) {
      v00 = fmaxf(v00, 0.0f); v01 = fmaxf(v01, 0.0f);
      v10 = fmaxf(v10, 0.0f); v11 = fmaxf(v11, 0.0f);
    }
    if (TRANS) {
      Ch[(size_t)cn0 * M + gm0] = (_Float16)v00;
      Ch[(size_t)cn1 * M + gm0] = (_Float16)v01;
      Ch[(size_t)cn0 * M + gm1] = (_Float16)v10;
      Ch[(size_t)cn1 * M + gm1] = (_Float16)v11;
    } else {
      if (Cf) {
        Cf[(size_t)gm0 * ldc + cn0] = v00; Cf[(size_t)gm0 * ldc + cn1] = v01;
        Cf[(size_t)gm1 * ldc + cn0] = v10; Cf[(size_t)gm1 * ldc + cn1] = v11;
      }
      if (Ch) {
        Ch[(size_t)gm0 * ldc + cn0] = (_Float16)v00; Ch[(size_t)gm0 * ldc + cn1] = (_Float16)v01;
        Ch[(size_t)gm1 * ldc + cn0] = (_Float16)v10; Ch[(size_t)gm1 * ldc + cn1] = (_Float16)v11;
      }
    }
  }
}

// ---------------------------------------------------------------------------
// Weight transpose + f32->f16:  out[N,K] = (f16) in[K,N]
// ---------------------------------------------------------------------------
__global__ void transpose_w_kernel(const float* __restrict__ in,
                                   _Float16* __restrict__ out, int Kd, int Nd)
{
  int idx = blockIdx.x * blockDim.x + threadIdx.x;
  if (idx >= Kd * Nd) return;
  int k = idx / Nd, n = idx - k * Nd;
  out[(size_t)n * Kd + k] = (_Float16)in[idx];
}

// ---------------------------------------------------------------------------
// LayerNorm of rows -> f16 (one 256-thread block per row, D==256)
// ---------------------------------------------------------------------------
__global__ __launch_bounds__(256)
void ln_rows_kernel(const float* __restrict__ x, const float* __restrict__ w,
                    const float* __restrict__ b, _Float16* __restrict__ xh)
{
  __shared__ float red[256];
  const int row = blockIdx.x, t = threadIdx.x;
  float v = x[(size_t)row * PD + t];
  red[t] = v; __syncthreads();
  for (int s = 128; s > 0; s >>= 1) { if (t < s) red[t] += red[t + s]; __syncthreads(); }
  const float mu = red[0] * (1.0f / PD); __syncthreads();
  const float d = v - mu;
  red[t] = d * d; __syncthreads();
  for (int s = 128; s > 0; s >>= 1) { if (t < s) red[t] += red[t + s]; __syncthreads(); }
  const float var = red[0] * (1.0f / PD);
  xh[(size_t)row * PD + t] = (_Float16)(d * rsqrtf(var + 1e-5f) * w[t] + b[t]);
}

// ---------------------------------------------------------------------------
// edges = mu + exp(logsigma)*noise ; e = LN(edges) -> f16
// ---------------------------------------------------------------------------
__global__ __launch_bounds__(256)
void edges_ln_kernel(const float* __restrict__ noise, const float* __restrict__ emu,
                     const float* __restrict__ els, const float* __restrict__ w,
                     const float* __restrict__ b, _Float16* __restrict__ eh)
{
  __shared__ float red[256];
  const int row = blockIdx.x, t = threadIdx.x;
  float v = emu[t] + __expf(els[t]) * noise[(size_t)row * PD + t];
  red[t] = v; __syncthreads();
  for (int s = 128; s > 0; s >>= 1) { if (t < s) red[t] += red[t + s]; __syncthreads(); }
  const float mu = red[0] * (1.0f / PD); __syncthreads();
  const float d = v - mu;
  red[t] = d * d; __syncthreads();
  for (int s = 128; s > 0; s >>= 1) { if (t < s) red[t] += red[t + s]; __syncthreads(); }
  const float var = red[0] * (1.0f / PD);
  eh[(size_t)row * PD + t] = (_Float16)(d * rsqrtf(var + 1e-5f) * w[t] + b[t]);
}

// ---------------------------------------------------------------------------
// Per-row softmax (+EPS, renorm) with top-k mask via radix select, row = 65536
// attn_h[s,n] = selected ? ((exp(d-max)/Z + EPS) / (1 + Ncols*EPS)) : 0
// ---------------------------------------------------------------------------
__global__ __launch_bounds__(256)
void attn_topk_kernel(const float* __restrict__ dots, _Float16* __restrict__ attn,
                      const int* __restrict__ kptr, int Ncols)
{
  __shared__ float fred[256];
  __shared__ unsigned hist[256];
  __shared__ int sbin, sacc;
  const int s = blockIdx.x, t = threadIdx.x;
  const float* row = dots + (size_t)s * Ncols;
  int kneed = *kptr;

  // row max
  float mx = -3.0e38f;
  for (int n = t; n < Ncols; n += 256) mx = fmaxf(mx, row[n]);
  fred[t] = mx; __syncthreads();
  for (int w = 128; w > 0; w >>= 1) { if (t < w) fred[t] = fmaxf(fred[t], fred[t + w]); __syncthreads(); }
  mx = fred[0]; __syncthreads();

  // partition function
  float z = 0.0f;
  for (int n = t; n < Ncols; n += 256) z += __expf(row[n] - mx);
  fred[t] = z; __syncthreads();
  for (int w = 128; w > 0; w >>= 1) { if (t < w) fred[t] += fred[t + w]; __syncthreads(); }
  z = fred[0]; __syncthreads();

  // radix select (descending) of the k-th largest key
  unsigned prefix = 0;
  for (int shift = 24; shift >= 0; shift -= 8) {
    hist[t] = 0u; __syncthreads();
    const unsigned himask = (shift == 24) ? 0u : (0xFFFFFFFFu << (shift + 8));
    for (int n = t; n < Ncols; n += 256) {
      unsigned u = fkey(row[n]);
      if ((u & himask) == prefix) atomicAdd(&hist[(u >> shift) & 255], 1u);
    }
    __syncthreads();
    if (t == 0) {
      int acc = 0, bsel = 0;
      for (int bb = 255; bb >= 0; --bb) {
        if (acc + (int)hist[bb] >= kneed) { bsel = bb; break; }
        acc += (int)hist[bb];
      }
      sbin = bsel; sacc = acc;
    }
    __syncthreads();
    prefix |= ((unsigned)sbin) << shift;
    kneed  -= sacc;
    __syncthreads();
  }

  const float invz   = 1.0f / z;
  const float renorm = 1.0f / (1.0f + (float)Ncols * PEPS);
  _Float16* arow = attn + (size_t)s * Ncols;
  for (int n = t; n < Ncols; n += 256) {
    const float d = row[n];
    const float val = (__expf(d - mx) * invz + PEPS) * renorm;
    arow[n] = (fkey(d) >= prefix) ? (_Float16)val : (_Float16)0.0f;
  }
}

// ---------------------------------------------------------------------------
// H = topk_mask(softmax(dots2, axis=1), k_e), row length 1024
// ---------------------------------------------------------------------------
__global__ __launch_bounds__(256)
void h_topk_kernel(const float* __restrict__ dots2, float* __restrict__ H,
                   const int* __restrict__ kptr)
{
  __shared__ float fred[256];
  __shared__ unsigned hist[256];
  __shared__ int sbin, sacc;
  const int s = blockIdx.x, t = threadIdx.x;
  const float* row = dots2 + (size_t)s * PNS;
  int kneed = *kptr;

  float vals[4];
  float mx = -3.0e38f;
  #pragma unroll
  for (int j = 0; j < 4; ++j) { vals[j] = row[t + j * 256]; mx = fmaxf(mx, vals[j]); }
  fred[t] = mx; __syncthreads();
  for (int w = 128; w > 0; w >>= 1) { if (t < w) fred[t] = fmaxf(fred[t], fred[t + w]); __syncthreads(); }
  mx = fred[0]; __syncthreads();

  float z = 0.0f;
  #pragma unroll
  for (int j = 0; j < 4; ++j) z += __expf(vals[j] - mx);
  fred[t] = z; __syncthreads();
  for (int w = 128; w > 0; w >>= 1) { if (t < w) fred[t] += fred[t + w]; __syncthreads(); }
  z = fred[0]; __syncthreads();

  unsigned prefix = 0;
  for (int shift = 24; shift >= 0; shift -= 8) {
    hist[t] = 0u; __syncthreads();
    const unsigned himask = (shift == 24) ? 0u : (0xFFFFFFFFu << (shift + 8));
    #pragma unroll
    for (int j = 0; j < 4; ++j) {
      unsigned u = fkey(vals[j]);
      if ((u & himask) == prefix) atomicAdd(&hist[(u >> shift) & 255], 1u);
    }
    __syncthreads();
    if (t == 0) {
      int acc = 0, bsel = 0;
      for (int bb = 255; bb >= 0; --bb) {
        if (acc + (int)hist[bb] >= kneed) { bsel = bb; break; }
        acc += (int)hist[bb];
      }
      sbin = bsel; sacc = acc;
    }
    __syncthreads();
    prefix |= ((unsigned)sbin) << shift;
    kneed  -= sacc;
    __syncthreads();
  }

  const float invz = 1.0f / z;
  float* hrow = H + (size_t)s * PNS;
  #pragma unroll
  for (int j = 0; j < 4; ++j) {
    const int n = t + j * 256;
    hrow[n] = (fkey(vals[j]) >= prefix) ? (__expf(vals[j] - mx) * invz) : 0.0f;
  }
}

// ---------------------------------------------------------------------------
// cat = [e, updates]  (f16, NS x 2D)
// ---------------------------------------------------------------------------
__global__ void concat_kernel(const _Float16* __restrict__ e,
                              const _Float16* __restrict__ upd,
                              _Float16* __restrict__ cat)
{
  int idx = blockIdx.x * blockDim.x + threadIdx.x;
  if (idx >= PNS * 2 * PD) return;
  int s = idx / (2 * PD), j = idx - s * (2 * PD);
  cat[idx] = (j < PD) ? e[(size_t)s * PD + j] : upd[(size_t)s * PD + (j - PD)];
}

// ---------------------------------------------------------------------------
extern "C" void kernel_launch(void* const* d_in, const int* in_sizes, int n_in,
                              void* d_out, int out_size, void* d_ws, size_t ws_size,
                              hipStream_t stream) {
  const float* inputs = (const float*)d_in[0];
  const float* noise  = (const float*)d_in[1];
  const float* emu    = (const float*)d_in[2];
  const float* els    = (const float*)d_in[3];
  const float* wq = (const float*)d_in[4];  const float* bq = (const float*)d_in[5];
  const float* wk = (const float*)d_in[6];  const float* bk = (const float*)d_in[7];
  const float* wv = (const float*)d_in[8];  const float* bv = (const float*)d_in[9];
  const float* w1 = (const float*)d_in[10]; const float* b1 = (const float*)d_in[11];
  const float* w2 = (const float*)d_in[12]; const float* b2 = (const float*)d_in[13];
  const float* ln_in_w = (const float*)d_in[14]; const float* ln_in_b = (const float*)d_in[15];
  const float* ln_e_w  = (const float*)d_in[16]; const float* ln_e_b  = (const float*)d_in[17];
  const int* k_n = (const int*)d_in[18];
  const int* k_e = (const int*)d_in[19];

  float* out_edges = (float*)d_out;                              // [NS, D]
  float* out_H     = out_edges + (size_t)PNS * PD;               // [N, NS]
  float* out_dots2 = out_H + (size_t)PN * PNS;                   // [N, NS]

  // workspace carve-up (256B aligned chunks)
  char* ws = (char*)d_ws; size_t off = 0;
  auto carve = [&](size_t bytes) -> void* {
    void* p = ws + off; off += (bytes + 255) & ~(size_t)255; return p;
  };
  _Float16* x_h    = (_Float16*)carve((size_t)PN * PD * 2);      // LN(inputs)
  _Float16* k_h    = (_Float16*)carve((size_t)PN * PD * 2);      // relu(x wk + bk)
  _Float16* vT_h   = (_Float16*)carve((size_t)PD * PN * 2);      // relu(x wv + bv)^T
  _Float16* q2_h   = (_Float16*)carve((size_t)PN * PD * 2);      // x wq + bq
  _Float16* e_h    = (_Float16*)carve((size_t)PNS * PD * 2);
  _Float16* q_h    = (_Float16*)carve((size_t)PNS * PD * 2);
  _Float16* upd_h  = (_Float16*)carve((size_t)PNS * PD * 2);
  _Float16* cat_h  = (_Float16*)carve((size_t)PNS * 2 * PD * 2);
  _Float16* hid_h  = (_Float16*)carve((size_t)PNS * PHID * 2);
  _Float16* edg_h  = (_Float16*)carve((size_t)PNS * PD * 2);
  _Float16* k2_h   = (_Float16*)carve((size_t)PNS * PD * 2);
  _Float16* wqT    = (_Float16*)carve((size_t)PD * PD * 2);
  _Float16* wkT    = (_Float16*)carve((size_t)PD * PD * 2);
  _Float16* wvT    = (_Float16*)carve((size_t)PD * PD * 2);
  _Float16* w1T    = (_Float16*)carve((size_t)PHID * 2 * PD * 2);
  _Float16* w2T    = (_Float16*)carve((size_t)PD * PHID * 2);
  float*    dots   = (float*)carve((size_t)PNS * PN * 4);        // 268 MB
  _Float16* attn_h = (_Float16*)carve((size_t)PNS * PN * 2);     // 134 MB

  const dim3 blk(256);

  // 1. pre-transpose weights to f16 [N,K]
  transpose_w_kernel<<<(PD * PD + 255) / 256, blk, 0, stream>>>(wq, wqT, PD, PD);
  transpose_w_kernel<<<(PD * PD + 255) / 256, blk, 0, stream>>>(wk, wkT, PD, PD);
  transpose_w_kernel<<<(PD * PD + 255) / 256, blk, 0, stream>>>(wv, wvT, PD, PD);
  transpose_w_kernel<<<(2 * PD * PHID + 255) / 256, blk, 0, stream>>>(w1, w1T, 2 * PD, PHID);
  transpose_w_kernel<<<(PHID * PD + 255) / 256, blk, 0, stream>>>(w2, w2T, PHID, PD);

  // 2. x = LN(inputs); e = LN(mu + exp(ls)*noise)
  ln_rows_kernel<<<PN, blk, 0, stream>>>(inputs, ln_in_w, ln_in_b, x_h);
  edges_ln_kernel<<<PNS, blk, 0, stream>>>(noise, emu, els, ln_e_w, ln_e_b, e_h);

  // 3. k = relu(x wk + bk) ; vT = relu(x wv + bv)^T ; q2 = x wq + bq
  gemm_nt_wmma<true,  false><<<dim3(PD / 128, PN / 64), blk, 0, stream>>>(
      x_h, wkT, bk, 1.0f, nullptr, k_h, PN, PD, PD, PD);
  gemm_nt_wmma<true,  true ><<<dim3(PD / 128, PN / 64), blk, 0, stream>>>(
      x_h, wvT, bv, 1.0f, nullptr, vT_h, PN, PD, PD, PD);
  gemm_nt_wmma<false, false><<<dim3(PD / 128, PN / 64), blk, 0, stream>>>(
      x_h, wqT, bq, 1.0f, nullptr, q2_h, PN, PD, PD, PD);

  // 4. q = relu(e wq + bq)
  gemm_nt_wmma<true, false><<<dim3(PD / 128, PNS / 64), blk, 0, stream>>>(
      e_h, wqT, bq, 1.0f, nullptr, q_h, PNS, PD, PD, PD);

  // 5. dots = SCALE * q @ k^T   [NS, N]
  gemm_nt_wmma<false, false><<<dim3(PN / 128, PNS / 64), blk, 0, stream>>>(
      q_h, k_h, nullptr, PSCALE, dots, nullptr, PNS, PN, PD, PN);

  // 6. attn = renorm(softmax(dots)+EPS) masked to top-k_n
  attn_topk_kernel<<<PNS, blk, 0, stream>>>(dots, attn_h, k_n, PN);

  // 7. updates = attn @ v   (NT with vT)   [NS, D]
  gemm_nt_wmma<false, false><<<dim3(PD / 128, PNS / 64), blk, 0, stream>>>(
      attn_h, vT_h, nullptr, 1.0f, nullptr, upd_h, PNS, PD, PN, PD);

  // 8. MLP: edges = relu([e, upd] w1 + b1) w2 + b2
  concat_kernel<<<(PNS * 2 * PD + 255) / 256, blk, 0, stream>>>(e_h, upd_h, cat_h);
  gemm_nt_wmma<true, false><<<dim3(PHID / 128, PNS / 64), blk, 0, stream>>>(
      cat_h, w1T, b1, 1.0f, nullptr, hid_h, PNS, PHID, 2 * PD, PHID);
  gemm_nt_wmma<false, false><<<dim3(PD / 128, PNS / 64), blk, 0, stream>>>(
      hid_h, w2T, b2, 1.0f, out_edges, edg_h, PNS, PD, PHID, PD);

  // 9. k2 = relu(edges wk + bk)
  gemm_nt_wmma<true, false><<<dim3(PD / 128, PNS / 64), blk, 0, stream>>>(
      edg_h, wkT, bk, 1.0f, nullptr, k2_h, PNS, PD, PD, PD);

  // 10. dots2 = SCALE * q2 @ k2^T   [N, NS]  (output)
  gemm_nt_wmma<false, false><<<dim3(PNS / 128, PN / 64), blk, 0, stream>>>(
      q2_h, k2_h, nullptr, PSCALE, out_dots2, nullptr, PN, PNS, PD, PNS);

  // 11. H = topk_mask(softmax(dots2), k_e)
  h_topk_kernel<<<PN, blk, 0, stream>>>(out_dots2, out_H, k_e);

  (void)in_sizes; (void)n_in; (void)out_size; (void)ws_size;
}
